// MambaEncoderLayer_77206332113509
// MI455X (gfx1250) — matrix operand order repaired and checked by reading, not compiled, not run on hardware
//
#include <hip/hip_runtime.h>
#include <stdint.h>

#define D_MODEL 1024
#define D_STATE 16
#define D_CONV  4
#define D_INNER 2048
#define DT_RANK 64
#define B_SZ    2
#define SEQ     1024
#define NTOK    (B_SZ*SEQ)
#define LN_EPS  1e-6f

typedef __attribute__((ext_vector_type(16))) __bf16   v16bf;
typedef __attribute__((ext_vector_type(8)))  float    v8f;
typedef __attribute__((ext_vector_type(8)))  uint32_t v8u;
typedef __attribute__((ext_vector_type(4)))  uint32_t v4u;

// ---------- f32 -> bf16 hi/lo split helpers (RNE, bit ops only) ----------
static __device__ __forceinline__ uint32_t f2bf(float f){
  uint32_t u = __builtin_bit_cast(uint32_t, f);
  return (u + 0x7fffu + ((u >> 16) & 1u)) >> 16;
}
static __device__ __forceinline__ float bf2f(uint32_t h){
  return __builtin_bit_cast(float, h << 16);
}
static __device__ __forceinline__ void split2(float f0, float f1, uint32_t &hi, uint32_t &lo){
  uint32_t h0 = f2bf(f0), h1 = f2bf(f1);
  float r0 = f0 - bf2f(h0), r1 = f1 - bf2f(h1);
  hi = h0 | (h1 << 16);
  lo = f2bf(r0) | (f2bf(r1) << 16);
}

// =========================================================================
// Pack A operand (row-major [R x C] f32, row stride ld) into WMMA A-fragment
// tiles. Tile block = 2048B: [0,1024) hi fragment, [1024,2048) lo fragment.
// Element (m,k): lane = (m&15)+16*((k>>3)&1), vgpr v = ((k&7)>>1)+4*(k>>4),
// byte = lane*32 + v*4 + (k&1)*2.
// =========================================================================
__global__ void pack_a_kernel(const float* __restrict__ src, int ld, int R, int C,
                              uint8_t* __restrict__ dst){
  int cpr = C >> 3;                       // 8-element chunks per row
  int t = blockIdx.x * blockDim.x + threadIdx.x;
  if (t >= R * cpr) return;
  int m  = t / cpr;
  int k8 = t - m * cpr;
  const float* s = src + (size_t)m * ld + k8 * 8;
  uint32_t hiw[4], low[4];
#pragma unroll
  for (int j = 0; j < 4; ++j) split2(s[2*j], s[2*j+1], hiw[j], low[j]);
  int ktiles = C >> 5;
  int tilem = m >> 4, tilek = k8 >> 2;
  int lane = (m & 15) + 16 * (k8 & 1);
  int vgrp = (k8 >> 1) & 1;
  size_t base = ((size_t)(tilem * ktiles + tilek) << 11) + (size_t)lane * 32 + vgrp * 16;
  v4u wh = {hiw[0], hiw[1], hiw[2], hiw[3]};
  v4u wl = {low[0], low[1], low[2], low[3]};
  *(v4u*)(dst + base)        = wh;
  *(v4u*)(dst + base + 1024) = wl;
}

// =========================================================================
// Pack B operand (row-major [K x N] f32, row stride ld) into WMMA B-fragment
// tiles. Element (k,n): lane = (n&15)+16*((k>>3)&1), v = ((k&7)>>1)+4*(k>>4) (within tile).
// Each thread handles 16 consecutive k of one column n (k 0-7 to lane n&15, k 8-15 to lane n&15+16).
// =========================================================================
__global__ void pack_b_kernel(const float* __restrict__ src, int ld, int K, int N,
                              uint8_t* __restrict__ dst){
  int t = blockIdx.x * blockDim.x + threadIdx.x;
  if (t >= (K >> 4) * N) return;
  int k16 = t / N;
  int n   = t - k16 * N;
  const float* s = src + (size_t)(k16 * 16) * ld + n;
  uint32_t hiw[8], low[8];
#pragma unroll
  for (int j = 0; j < 8; ++j)
    split2(s[(size_t)(2*j) * ld], s[(size_t)(2*j+1) * ld], hiw[j], low[j]);
  int ntiles = N >> 4;
  int tilek = k16 >> 1, tilen = n >> 4;
  int lane = (n & 15);
  size_t base = ((size_t)(tilek * ntiles + tilen) << 11) + (size_t)lane * 32 + 16 * (k16 & 1);
  v4u wh0 = {hiw[0], hiw[1], hiw[2], hiw[3]};
  v4u wh1 = {hiw[4], hiw[5], hiw[6], hiw[7]};
  v4u wl0 = {low[0], low[1], low[2], low[3]};
  v4u wl1 = {low[4], low[5], low[6], low[7]};
  *(v4u*)(dst + base)            = wh0;
  *(v4u*)(dst + base + 512)      = wh1;
  *(v4u*)(dst + base + 1024)     = wl0;
  *(v4u*)(dst + base + 1024+512) = wl1;
}

// =========================================================================
// bf16x2-split WMMA GEMM: C[M,N] f32 = A[M,K] x B[K,N], operands in fragment-
// packed layouts, L2-resident (192MB L2 on MI455X) so no LDS staging.
// Block = 256 threads (8 wave32), block tile 128x64, wave tile 32x32 =
// 2x2 WMMA tiles (small enough that fragments + acc fit without spills).
// 3 WMMAs per tile-pair per K-step implement the bf16 hi/lo split.
// Out-of-range B tiles (only GEMM3, N=96) are CLAMPED to the last valid tile
// so the K-loop is branch-free; only the epilogue store is masked.
// =========================================================================
__global__ void gemm_wmma_kernel(const uint8_t* __restrict__ Apk,
                                 const uint8_t* __restrict__ Bpk,
                                 float* __restrict__ C,
                                 int N, int K){
  const int KT = K >> 5;
  const int n_tiles = N >> 4;
  int tid = threadIdx.x;
  int lane = tid & 31;
  int wid  = tid >> 5;
  int wm = wid >> 1;                 // 0..3
  int wn = wid & 1;                  // 0..1
  int mtile0 = blockIdx.x * 8 + wm * 2;
  int ntile0 = blockIdx.y * 4 + wn * 2;
  bool nv0 = (ntile0 + 0) < n_tiles;
  bool nv1 = (ntile0 + 1) < n_tiles;
  // Clamped tile indices: loads always touch valid memory; stores are masked.
  int t0 = nv0 ? (ntile0 + 0) : (n_tiles - 1);
  int t1 = nv1 ? (ntile0 + 1) : (n_tiles - 1);

  v8f acc00, acc01, acc10, acc11;
  {
    v8f z = {0.f,0.f,0.f,0.f,0.f,0.f,0.f,0.f};
    acc00 = z; acc01 = z; acc10 = z; acc11 = z;
  }

  // Wave-uniform tile bases; per-lane part is loff.
  const uint8_t* pa0 = Apk + (((size_t)(mtile0 + 0) * KT) << 11);
  const uint8_t* pa1 = Apk + (((size_t)(mtile0 + 1) * KT) << 11);
  const uint8_t* pb0 = Bpk + (((size_t)t0) << 11);
  const uint8_t* pb1 = Bpk + (((size_t)t1) << 11);
  const uint32_t loff = (uint32_t)lane * 32u;
  const size_t bstep = ((size_t)n_tiles) << 11;

  size_t aoff = loff;          // advances by 2048 per kt (A tiles contiguous in K)
  size_t boff = loff;          // advances by n_tiles*2048 per kt

  for (int kt = 0; kt < KT; ++kt){
    v16bf bh0 = __builtin_bit_cast(v16bf, *(const v8u*)(pb0 + boff));
    v16bf bl0 = __builtin_bit_cast(v16bf, *(const v8u*)(pb0 + boff + 1024));
    v16bf bh1 = __builtin_bit_cast(v16bf, *(const v8u*)(pb1 + boff));
    v16bf bl1 = __builtin_bit_cast(v16bf, *(const v8u*)(pb1 + boff + 1024));

    {
      v16bf ah = __builtin_bit_cast(v16bf, *(const v8u*)(pa0 + aoff));
      v16bf al = __builtin_bit_cast(v16bf, *(const v8u*)(pa0 + aoff + 1024));
      acc00 = __builtin_amdgcn_wmma_f32_16x16x32_bf16(false, ah, false, bh0, (short)0, acc00, false, false);
      acc00 = __builtin_amdgcn_wmma_f32_16x16x32_bf16(false, ah, false, bl0, (short)0, acc00, false, false);
      acc00 = __builtin_amdgcn_wmma_f32_16x16x32_bf16(false, al, false, bh0, (short)0, acc00, false, false);
      acc01 = __builtin_amdgcn_wmma_f32_16x16x32_bf16(false, ah, false, bh1, (short)0, acc01, false, false);
      acc01 = __builtin_amdgcn_wmma_f32_16x16x32_bf16(false, ah, false, bl1, (short)0, acc01, false, false);
      acc01 = __builtin_amdgcn_wmma_f32_16x16x32_bf16(false, al, false, bh1, (short)0, acc01, false, false);
    }
    {
      v16bf ah = __builtin_bit_cast(v16bf, *(const v8u*)(pa1 + aoff));
      v16bf al = __builtin_bit_cast(v16bf, *(const v8u*)(pa1 + aoff + 1024));
      acc10 = __builtin_amdgcn_wmma_f32_16x16x32_bf16(false, ah, false, bh0, (short)0, acc10, false, false);
      acc10 = __builtin_amdgcn_wmma_f32_16x16x32_bf16(false, ah, false, bl0, (short)0, acc10, false, false);
      acc10 = __builtin_amdgcn_wmma_f32_16x16x32_bf16(false, al, false, bh0, (short)0, acc10, false, false);
      acc11 = __builtin_amdgcn_wmma_f32_16x16x32_bf16(false, ah, false, bh1, (short)0, acc11, false, false);
      acc11 = __builtin_amdgcn_wmma_f32_16x16x32_bf16(false, ah, false, bl1, (short)0, acc11, false, false);
      acc11 = __builtin_amdgcn_wmma_f32_16x16x32_bf16(false, al, false, bh1, (short)0, acc11, false, false);
    }

    aoff += 2048;
    boff += bstep;
  }

  // C/D layout: VGPR r -> m = r + 8*(lane>>4), n = lane&15
  int radd = (lane >> 4) * 8;
  int ncol = lane & 15;
  {
    int row0 = (mtile0 + 0) * 16;
    if (nv0){
      int col = (ntile0 + 0) * 16 + ncol;
#pragma unroll
      for (int r = 0; r < 8; ++r) C[(size_t)(row0 + r + radd) * N + col] = acc00[r];
    }
    if (nv1){
      int col = (ntile0 + 1) * 16 + ncol;
#pragma unroll
      for (int r = 0; r < 8; ++r) C[(size_t)(row0 + r + radd) * N + col] = acc01[r];
    }
  }
  {
    int row0 = (mtile0 + 1) * 16;
    if (nv0){
      int col = (ntile0 + 0) * 16 + ncol;
#pragma unroll
      for (int r = 0; r < 8; ++r) C[(size_t)(row0 + r + radd) * N + col] = acc10[r];
    }
    if (nv1){
      int col = (ntile0 + 1) * 16 + ncol;
#pragma unroll
      for (int r = 0; r < 8; ++r) C[(size_t)(row0 + r + radd) * N + col] = acc11[r];
    }
  }
}

// =========================================================================
// Depthwise causal conv (D_CONV=4) + SiLU. Reads xi = xz[:, :2048].
// Writes xc as f32 (for the scan) and as A-fragment-packed hi/lo (for GEMM3).
// Each thread handles 8 consecutive channels of one token.
// =========================================================================
__global__ void conv_silu_kernel(const float* __restrict__ xz,
                                 const float* __restrict__ conv_w,
                                 const float* __restrict__ conv_b,
                                 float* __restrict__ xcf,
                                 uint8_t* __restrict__ xcpk){
  int t = blockIdx.x * blockDim.x + threadIdx.x;
  const int cpr = D_INNER >> 3;            // 256
  int m = t / cpr;
  if (m >= NTOK) return;
  int d8 = t - m * cpr;
  int d0 = d8 * 8;
  int b = m >> 10, l = m & 1023;
  float acc[8];
#pragma unroll
  for (int u = 0; u < 8; ++u) acc[u] = conv_b[d0 + u];
#pragma unroll
  for (int k = 0; k < D_CONV; ++k){
    int ls = l - (D_CONV - 1) + k;
    if (ls < 0) continue;
    const float* row = xz + (size_t)(b * SEQ + ls) * (2 * D_INNER) + d0;
#pragma unroll
    for (int u = 0; u < 8; ++u) acc[u] += row[u] * conv_w[(d0 + u) * D_CONV + k];
  }
#pragma unroll
  for (int u = 0; u < 8; ++u){
    float v = acc[u];
    acc[u] = v / (1.f + expf(-v));         // SiLU
  }
  float* o = xcf + (size_t)m * D_INNER + d0;
#pragma unroll
  for (int u = 0; u < 8; ++u) o[u] = acc[u];

  uint32_t hiw[4], low[4];
#pragma unroll
  for (int j = 0; j < 4; ++j) split2(acc[2*j], acc[2*j+1], hiw[j], low[j]);
  const int ktiles = D_INNER >> 5;
  int tilem = m >> 4, tilek = d8 >> 2;
  int lane = (m & 15) + 16 * (d8 & 1);
  int vgrp = (d8 >> 1) & 1;
  size_t base = ((size_t)(tilem * ktiles + tilek) << 11) + (size_t)lane * 32 + vgrp * 16;
  v4u wh = {hiw[0], hiw[1], hiw[2], hiw[3]};
  v4u wl = {low[0], low[1], low[2], low[3]};
  *(v4u*)(xcpk + base)        = wh;
  *(v4u*)(xcpk + base + 1024) = wl;
}

// =========================================================================
// Selective scan. 16 lanes per (b,d) pair, one state n per lane, sequential
// over L. Cross-state contraction h.C via wave32 __shfl_xor (width 16).
// Applies softplus(dt+b_dt), +u*D, *silu(z); writes y fragment-packed.
// =========================================================================
__global__ void scan_kernel(const float* __restrict__ dt_pre,
                            const float* __restrict__ b_dt,
                            const float* __restrict__ A_log,
                            const float* __restrict__ x_dbl,
                            const float* __restrict__ xcf,
                            const float* __restrict__ D_param,
                            const float* __restrict__ xz,
                            uint8_t* __restrict__ ypk){
  int t = blockIdx.x * blockDim.x + threadIdx.x;
  int g = t >> 4;
  int n = t & 15;
  if (g >= B_SZ * D_INNER) return;
  int b = g >> 11;                          // /2048
  int d = g & (D_INNER - 1);
  float Ac  = -expf(A_log[d * D_STATE + n]);
  float Dp  = D_param[d];
  float bdt = b_dt[d];
  float h = 0.f;

  const int ktiles = D_INNER >> 5;
  int lane_y = 0, vby = 0;
  if (n == 0){
    lane_y = 16 * ((d >> 3) & 1);
    vby = (((d & 7) >> 1) + 4 * ((d >> 4) & 1)) * 4 + (d & 1) * 2;
  }

  for (int l = 0; l < SEQ; ++l){
    int m = b * SEQ + l;
    float xdt = dt_pre[(size_t)m * D_INNER + d] + bdt;
    float dt = (xdt > 20.f) ? xdt : log1pf(expf(xdt));   // softplus
    float u  = xcf[(size_t)m * D_INNER + d];
    float Bv = x_dbl[(size_t)m * 96 + DT_RANK + n];
    float Cv = x_dbl[(size_t)m * 96 + DT_RANK + D_STATE + n];
    h = expf(dt * Ac) * h + dt * Bv * u;
    float p = h * Cv;
    p += __shfl_xor(p, 1, 16);
    p += __shfl_xor(p, 2, 16);
    p += __shfl_xor(p, 4, 16);
    p += __shfl_xor(p, 8, 16);
    if (n == 0){
      float z = xz[(size_t)m * (2 * D_INNER) + D_INNER + d];
      float y = (p + u * Dp) * (z / (1.f + expf(-z)));
      uint32_t hy = f2bf(y);
      uint32_t ly = f2bf(y - bf2f(hy));
      size_t base = ((size_t)((m >> 4) * ktiles + (d >> 5)) << 11)
                  + (size_t)((m & 15) + lane_y) * 32 + vby;
      *(uint16_t*)(ypk + base)        = (uint16_t)hy;
      *(uint16_t*)(ypk + base + 1024) = (uint16_t)ly;
    }
  }
}

// =========================================================================
// LayerNorm + residual: out = x + LN(m)*gamma + beta. One block per token.
// =========================================================================
__global__ void ln_residual_kernel(const float* __restrict__ mbuf,
                                   const float* __restrict__ x,
                                   const float* __restrict__ gamma,
                                   const float* __restrict__ beta,
                                   float* __restrict__ out){
  __shared__ float red[2][8];
  int m = blockIdx.x;
  int tid = threadIdx.x;
  const float* row = mbuf + (size_t)m * D_MODEL;
  float s = 0.f, s2 = 0.f;
  for (int c = tid; c < D_MODEL; c += 256){ float v = row[c]; s += v; s2 += v * v; }
#pragma unroll
  for (int off = 16; off >= 1; off >>= 1){
    s  += __shfl_xor(s,  off, 32);
    s2 += __shfl_xor(s2, off, 32);
  }
  int wid = tid >> 5, lane = tid & 31;
  if (lane == 0){ red[0][wid] = s; red[1][wid] = s2; }
  __syncthreads();
  s = 0.f; s2 = 0.f;
#pragma unroll
  for (int w = 0; w < 8; ++w){ s += red[0][w]; s2 += red[1][w]; }
  float mu = s * (1.f / D_MODEL);
  float var = s2 * (1.f / D_MODEL) - mu * mu;
  float rs = rsqrtf(var + LN_EPS);
  for (int c = tid; c < D_MODEL; c += 256){
    float v = (row[c] - mu) * rs * gamma[c] + beta[c];
    out[(size_t)m * D_MODEL + c] = x[(size_t)m * D_MODEL + c] + v;
  }
}

// =========================================================================
extern "C" void kernel_launch(void* const* d_in, const int* in_sizes, int n_in,
                              void* d_out, int out_size, void* d_ws, size_t ws_size,
                              hipStream_t stream){
  const float* x       = (const float*)d_in[0];
  const float* W_in    = (const float*)d_in[1];
  const float* conv_w  = (const float*)d_in[2];
  const float* conv_b  = (const float*)d_in[3];
  const float* W_xproj = (const float*)d_in[4];
  const float* W_dt    = (const float*)d_in[5];
  const float* b_dt    = (const float*)d_in[6];
  const float* A_log   = (const float*)d_in[7];
  const float* D_param = (const float*)d_in[8];
  const float* W_out   = (const float*)d_in[9];
  const float* gamma   = (const float*)d_in[10];
  const float* beta    = (const float*)d_in[11];
  float* out = (float*)d_out;

  uint8_t* ws = (uint8_t*)d_ws;
  size_t off = 0;
  auto alloc = [&](size_t bytes)->uint8_t*{
    uint8_t* p = ws + off;
    off += (bytes + 255) & ~(size_t)255;
    return p;
  };

  float*   xz    = (float*)  alloc((size_t)NTOK * 4096 * 4);   // 32 MB
  float*   xcf   = (float*)  alloc((size_t)NTOK * 2048 * 4);   // 16 MB
  uint8_t* xcpk  =           alloc((size_t)NTOK * 2048 * 4);   // 16 MB
  float*   xdbl  = (float*)  alloc((size_t)NTOK * 96   * 4);
  float*   dtpre = (float*)  alloc((size_t)NTOK * 2048 * 4);   // 16 MB
  uint8_t* wxpk  =           alloc((size_t)2048 * 96   * 4);
  uint8_t* dtrpk =           alloc((size_t)NTOK * 64   * 4);
  uint8_t* wdpk  =           alloc((size_t)64   * 2048 * 4);
  float*   mbuf  = (float*)  alloc((size_t)NTOK * 1024 * 4);   // 8 MB
  uint8_t* regA  =           alloc((size_t)25165824);          // 24 MB reused
  uint8_t* xpk  = regA;                    // 8 MB   (live until GEMM1)
  uint8_t* wipk = regA + 8388608;          // 16 MB  (live until GEMM1)
  uint8_t* ypk  = regA;                    // 16 MB  (written by scan, after GEMM1)
  uint8_t* wopk = regA + 16777216;         // 8 MB   (packed after GEMM1)
  (void)in_sizes; (void)n_in; (void)out_size; (void)ws_size;

  // --- pack inputs for GEMM1 ---
  pack_a_kernel<<<(NTOK * (D_MODEL/8)) / 256, 256, 0, stream>>>(x, D_MODEL, NTOK, D_MODEL, xpk);
  pack_b_kernel<<<((D_MODEL/16) * 4096) / 256, 256, 0, stream>>>(W_in, 4096, D_MODEL, 4096, wipk);
  // --- GEMM1: xz = x @ W_in  (2048 x 1024 x 4096) ---
  gemm_wmma_kernel<<<dim3(NTOK/128, 4096/64), 256, 0, stream>>>(xpk, wipk, xz, 4096, D_MODEL);
  // --- conv + SiLU -> xc (f32 + packed) ---
  conv_silu_kernel<<<(NTOK * (D_INNER/8)) / 256, 256, 0, stream>>>(xz, conv_w, conv_b, xcf, xcpk);
  // --- GEMM3: x_dbl = xc @ W_xproj  (2048 x 2048 x 96) ---
  pack_b_kernel<<<(((D_INNER/16) * 96) + 255) / 256, 256, 0, stream>>>(W_xproj, 96, D_INNER, 96, wxpk);
  gemm_wmma_kernel<<<dim3(NTOK/128, 2), 256, 0, stream>>>(xcpk, wxpk, xdbl, 96, D_INNER);
  // --- GEMM4: dt_pre = dt_r @ W_dt  (2048 x 64 x 2048) ---
  pack_a_kernel<<<(NTOK * (DT_RANK/8)) / 256, 256, 0, stream>>>(xdbl, 96, NTOK, DT_RANK, dtrpk);
  pack_b_kernel<<<((DT_RANK/16) * D_INNER) / 256, 256, 0, stream>>>(W_dt, D_INNER, DT_RANK, D_INNER, wdpk);
  gemm_wmma_kernel<<<dim3(NTOK/128, D_INNER/64), 256, 0, stream>>>(dtrpk, wdpk, dtpre, D_INNER, DT_RANK);
  // --- selective scan -> y (fragment-packed) ---
  scan_kernel<<<(B_SZ * D_INNER * 16) / 256, 256, 0, stream>>>(dtpre, b_dt, A_log, xdbl, xcf, D_param, xz, ypk);
  // --- GEMM2: m = y @ W_out  (2048 x 2048 x 1024) ---
  pack_b_kernel<<<((D_INNER/16) * D_MODEL) / 256, 256, 0, stream>>>(W_out, D_MODEL, D_INNER, D_MODEL, wopk);
  gemm_wmma_kernel<<<dim3(NTOK/128, D_MODEL/64), 256, 0, stream>>>(ypk, wopk, mbuf, D_MODEL, D_INNER);
  // --- LN + residual ---
  ln_residual_kernel<<<NTOK, 256, 0, stream>>>(mbuf, x, gamma, beta, out);
}
